// TopM_MHSA_5686536699992
// MI455X (gfx1250) — matrix operand order repaired and checked by reading, not compiled, not run on hardware
//
#include <hip/hip_runtime.h>
#include <hip/hip_bf16.h>

// ---- problem constants ----
#define LAYERS 2
#define BATCH  8
#define NTOK   1024
#define CDIM   512
#define HEADS  8
#define DHEAD  64
#define FDIM   2048
#define TOPM   64
#define BN     (BATCH * NTOK)   // 8192
#define SCALE  0.125f           // DHEAD^-0.5

typedef __bf16 bf16;
typedef __attribute__((ext_vector_type(16))) __bf16 bf16x16;
typedef __attribute__((ext_vector_type(8)))  __bf16 bf16x8;
typedef __attribute__((ext_vector_type(8)))  float   f32x8;

// -------- WMMA helpers --------
__device__ __forceinline__ f32x8 wmma_bf16(bf16x16 a, bf16x16 b, f32x8 c) {
    return __builtin_amdgcn_wmma_f32_16x16x32_bf16(false, a, false, b, (short)0, c,
                                                   false, false);
}

// A-matrix 16x32 bf16 fragment: lane&15 selects the row (caller passes the row
// pointer); lanes 0-15 hold K = [0..7]+[16..23], lanes 16-31 hold K = [8..15]+[24..31].
__device__ __forceinline__ bf16x16 load_a_frag(const bf16* __restrict__ row, int kk, int lane) {
    int koff = kk + ((lane >> 4) << 3);
    bf16x8 lo = *(const bf16x8*)(row + koff);
    bf16x8 hi = *(const bf16x8*)(row + koff + 16);
    bf16x16 f;
#pragma unroll
    for (int i = 0; i < 8; ++i) { f[i] = lo[i]; f[i + 8] = hi[i]; }
    return f;
}

// B-matrix 32x16 bf16 fragment: lane&15 selects the column (caller passes the
// column-major row pointer, i.e. W^T row); lanes 0-15 hold K = [0..7]+[16..23],
// lanes 16-31 hold K = [8..15]+[24..31] (same order as the A fragment).
__device__ __forceinline__ bf16x16 load_b_frag(const bf16* __restrict__ colrow, int kk, int lane) {
    int koff = kk + ((lane >> 4) << 3);
    bf16x8 lo = *(const bf16x8*)(colrow + koff);
    bf16x8 hi = *(const bf16x8*)(colrow + koff + 16);
    bf16x16 f;
#pragma unroll
    for (int i = 0; i < 8; ++i) { f[i] = lo[i]; f[i + 8] = hi[i]; }
    return f;
}

__device__ __forceinline__ float unflip(unsigned v) {
    return __uint_as_float((v & 0x80000000u) ? (v & 0x7FFFFFFFu) : ~v);
}

// -------- elementwise: out = x + pos --------
__global__ void __launch_bounds__(256) k_add_pos(const float* __restrict__ x,
                                                 const float* __restrict__ pos,
                                                 float* __restrict__ out) {
    int i = blockIdx.x * 256 + threadIdx.x;              // float4 index
    float4 a = ((const float4*)x)[i];
    float4 p = ((const float4*)pos)[i];
    float4 r; r.x = a.x + p.x; r.y = a.y + p.y; r.z = a.z + p.z; r.w = a.w + p.w;
    ((float4*)out)[i] = r;
}

// -------- LayerNorm (fp32 in) -> bf16 out; one wave per row of 512 --------
__global__ void __launch_bounds__(128) k_ln(const float* __restrict__ xin,
                                            const float* __restrict__ g,
                                            const float* __restrict__ bta,
                                            bf16* __restrict__ out) {
    int row  = blockIdx.x * 4 + (threadIdx.x >> 5);
    int lane = threadIdx.x & 31;
    const float* xr = xin + (size_t)row * CDIM;
    float v[16];
    float s = 0.f;
#pragma unroll
    for (int i = 0; i < 4; ++i) {
        float4 t = *(const float4*)(xr + lane * 16 + i * 4);
        v[i*4+0] = t.x; v[i*4+1] = t.y; v[i*4+2] = t.z; v[i*4+3] = t.w;
        s += t.x + t.y + t.z + t.w;
    }
#pragma unroll
    for (int off = 16; off; off >>= 1) s += __shfl_xor(s, off, 32);
    float mean = s * (1.f / CDIM);
    float qv = 0.f;
#pragma unroll
    for (int i = 0; i < 16; ++i) { float d = v[i] - mean; qv += d * d; }
#pragma unroll
    for (int off = 16; off; off >>= 1) qv += __shfl_xor(qv, off, 32);
    float rstd = rsqrtf(qv * (1.f / CDIM) + 1e-5f);
    bf16* orow = out + (size_t)row * CDIM;
#pragma unroll
    for (int ch = 0; ch < 2; ++ch) {
        bf16x8 pk;
#pragma unroll
        for (int i = 0; i < 8; ++i) {
            int c = lane * 16 + ch * 8 + i;
            pk[i] = (bf16)((v[ch * 8 + i] - mean) * rstd * g[c] + bta[c]);
        }
        *(bf16x8*)(orow + lane * 16 + ch * 8) = pk;
    }
}

// -------- transpose+convert: wt[n][k] = (bf16) w[k][n] --------
__global__ void __launch_bounds__(256) k_wt(const float* __restrict__ w,
                                            bf16* __restrict__ wt,
                                            int K, int Ncols) {
    int idx = blockIdx.x * 256 + threadIdx.x;
    if (idx >= Ncols * K) return;
    int n = idx / K, kk = idx - n * K;
    wt[idx] = (bf16)w[(size_t)kk * Ncols + n];
}

// -------- generic WMMA GEMM: out = A[M,K](bf16) * WT[N,K](bf16) + bias --------
// Each wave computes a 32x64 tile (2 A-frags x 4 B-frags = 8 WMMAs / K-step);
// block = 8 waves -> 64x256 tile.  MODE 0: fp32 residual add; MODE 1: GELU->bf16.
template <int MODE>
__global__ void __launch_bounds__(256) k_gemm(const bf16* __restrict__ A,
                                              const bf16* __restrict__ WT,
                                              const float* __restrict__ bias,
                                              void* __restrict__ outv,
                                              int K, int Ncols) {
    int lane = threadIdx.x & 31, wid = threadIdx.x >> 5;
    int row0 = blockIdx.y * 64 + (wid >> 2) * 32;
    int col0 = blockIdx.x * 256 + (wid & 3) * 64;
    const bf16* Arow0 = A + (size_t)(row0 + (lane & 15)) * K;
    const bf16* Arow1 = Arow0 + (size_t)16 * K;

    f32x8 acc[2][4];
#pragma unroll
    for (int m = 0; m < 2; ++m)
#pragma unroll
        for (int t = 0; t < 4; ++t)
#pragma unroll
            for (int i = 0; i < 8; ++i) acc[m][t][i] = 0.f;

    for (int kk = 0; kk < K; kk += 32) {
        __builtin_prefetch(Arow0 + kk + 64, 0, 0);
        bf16x16 a0 = load_a_frag(Arow0, kk, lane);
        bf16x16 a1 = load_a_frag(Arow1, kk, lane);
#pragma unroll
        for (int t = 0; t < 4; ++t) {
            const bf16* Brow = WT + (size_t)(col0 + t * 16 + (lane & 15)) * K;
            bf16x16 b = load_b_frag(Brow, kk, lane);
            acc[0][t] = wmma_bf16(a0, b, acc[0][t]);
            acc[1][t] = wmma_bf16(a1, b, acc[1][t]);
        }
    }

#pragma unroll
    for (int m = 0; m < 2; ++m) {
        int rbase = row0 + m * 16 + ((lane >> 4) << 3);
#pragma unroll
        for (int t = 0; t < 4; ++t) {
            int col = col0 + t * 16 + (lane & 15);
            float bb = bias[col];
#pragma unroll
            for (int j = 0; j < 8; ++j) {
                float val = acc[m][t][j] + bb;
                size_t idx = (size_t)(rbase + j) * Ncols + col;
                if (MODE == 0) {
                    float* o = (float*)outv;
                    o[idx] += val;
                } else {
                    bf16* o = (bf16*)outv;
                    float ge = 0.5f * val * (1.f + erff(val * 0.70710678118654752f));
                    o[idx] = (bf16)ge;
                }
            }
        }
    }
}

// -------- QKV GEMM (32x64/wave) with head-major scatter epilogue --------
__global__ void __launch_bounds__(256) k_gemm_qkv(const bf16* __restrict__ A,
                                                  const bf16* __restrict__ WT,
                                                  const float* __restrict__ bias,
                                                  bf16* __restrict__ qo,
                                                  bf16* __restrict__ ko,
                                                  bf16* __restrict__ vT) {
    const int K = CDIM;
    int lane = threadIdx.x & 31, wid = threadIdx.x >> 5;
    int row0 = blockIdx.y * 64 + (wid >> 2) * 32;
    int col0 = blockIdx.x * 256 + (wid & 3) * 64;
    const bf16* Arow0 = A + (size_t)(row0 + (lane & 15)) * K;
    const bf16* Arow1 = Arow0 + (size_t)16 * K;

    f32x8 acc[2][4];
#pragma unroll
    for (int m = 0; m < 2; ++m)
#pragma unroll
        for (int t = 0; t < 4; ++t)
#pragma unroll
            for (int i = 0; i < 8; ++i) acc[m][t][i] = 0.f;

    for (int kk = 0; kk < K; kk += 32) {
        bf16x16 a0 = load_a_frag(Arow0, kk, lane);
        bf16x16 a1 = load_a_frag(Arow1, kk, lane);
#pragma unroll
        for (int t = 0; t < 4; ++t) {
            const bf16* Brow = WT + (size_t)(col0 + t * 16 + (lane & 15)) * K;
            bf16x16 b = load_b_frag(Brow, kk, lane);
            acc[0][t] = wmma_bf16(a0, b, acc[0][t]);
            acc[1][t] = wmma_bf16(a1, b, acc[1][t]);
        }
    }

#pragma unroll
    for (int m = 0; m < 2; ++m) {
        int rbase = row0 + m * 16 + ((lane >> 4) << 3);
#pragma unroll
        for (int t = 0; t < 4; ++t) {
            int col = col0 + t * 16 + (lane & 15);
            float bb = bias[col];
            int which = col / CDIM;             // 0=q 1=k 2=v
            int c = col - which * CDIM;
            int h = c >> 6, d = c & 63;
#pragma unroll
            for (int j = 0; j < 8; ++j) {
                int row = rbase + j;
                int bI = row >> 10, n = row & 1023;
                float val = acc[m][t][j] + bb;
                size_t bh = (size_t)(bI * HEADS + h);
                if (which == 0)      qo[(bh * NTOK + n) * DHEAD + d] = (bf16)(val * SCALE);
                else if (which == 1) ko[(bh * NTOK + n) * DHEAD + d] = (bf16)val;
                else                 vT[(bh * DHEAD + d) * NTOK + n] = (bf16)val;
            }
        }
    }
}

// -------- Top-M attention: one block per (b, h, 16-query tile) --------
__global__ void __launch_bounds__(256) k_attn(const bf16* __restrict__ q,
                                              const bf16* __restrict__ kbuf,
                                              const bf16* __restrict__ vT,
                                              bf16* __restrict__ o) {
    __shared__ float S[16][NTOK];       // 64 KB score tile
    __shared__ float Oacc[16 * DHEAD];  // 4 KB output accumulator
    int qt = blockIdx.x, h = blockIdx.y, b = blockIdx.z;
    int tid = threadIdx.x, lane = tid & 31, w = tid >> 5;
    size_t bh = (size_t)(b * HEADS + h);
    const bf16* qb = q    + bh * NTOK * DHEAD;
    const bf16* kb = kbuf + bh * NTOK * DHEAD;
    const bf16* vb = vT   + bh * DHEAD * NTOK;

    for (int i = tid; i < 16 * DHEAD; i += 256) Oacc[i] = 0.f;

    // ---- phase 1: S = (Q*SCALE) K^T, wave w covers keys [w*128, w*128+128) ----
    {
        int qrow = qt * 16 + (lane & 15);
        const bf16* qrowp = qb + (size_t)qrow * DHEAD;
        bf16x16 a0 = load_a_frag(qrowp, 0, lane);
        bf16x16 a1 = load_a_frag(qrowp, 32, lane);
#pragma unroll
        for (int t = 0; t < 8; ++t) {
            int key = w * 128 + t * 16 + (lane & 15);
            const bf16* krow = kb + (size_t)key * DHEAD;
            f32x8 acc;
#pragma unroll
            for (int i = 0; i < 8; ++i) acc[i] = 0.f;
            acc = wmma_bf16(a0, load_b_frag(krow, 0, lane), acc);
            acc = wmma_bf16(a1, load_b_frag(krow, 32, lane), acc);
            int rb = (lane >> 4) << 3;
#pragma unroll
            for (int j = 0; j < 8; ++j) S[rb + j][key] = acc[j];
        }
    }
    __syncthreads();

    // ---- phase 2: exact top-M threshold (32-bit radix select) + masked softmax ----
    {
        int row = tid >> 4, sub = tid & 15;    // 16 threads per row
        unsigned fk[64];
#pragma unroll
        for (int i = 0; i < 64; ++i) {
            unsigned u = __float_as_uint(S[row][sub + (i << 4)]);
            fk[i] = (u & 0x80000000u) ? ~u : (u | 0x80000000u);
        }
        unsigned prefix = 0;
        int need = TOPM;
        for (int bit = 31; bit >= 0; --bit) {
            unsigned hm = (bit == 31) ? 0u : (0xFFFFFFFFu << (bit + 1));
            unsigned cb = 1u << bit;
            int cnt = 0;
#pragma unroll
            for (int i = 0; i < 64; ++i)
                cnt += (((fk[i] & hm) == prefix) && (fk[i] & cb)) ? 1 : 0;
#pragma unroll
            for (int off = 8; off; off >>= 1) cnt += __shfl_xor(cnt, off, 32);
            if (cnt >= need) prefix |= cb; else need -= cnt;
        }
        float thr = unflip(prefix);            // M-th largest score of this row
        float mx = -3.4e38f;
#pragma unroll
        for (int i = 0; i < 64; ++i) mx = fmaxf(mx, unflip(fk[i]));
#pragma unroll
        for (int off = 8; off; off >>= 1) mx = fmaxf(mx, __shfl_xor(mx, off, 32));
        float sum = 0.f;
#pragma unroll
        for (int i = 0; i < 64; ++i) {
            float sv = unflip(fk[i]);
            float e = (sv >= thr) ? __expf(sv - mx) : 0.f;
            S[row][sub + (i << 4)] = e;
            sum += e;
        }
#pragma unroll
        for (int off = 8; off; off >>= 1) sum += __shfl_xor(sum, off, 32);
        float inv = 1.f / sum;
#pragma unroll
        for (int i = 0; i < 64; ++i) S[row][sub + (i << 4)] *= inv;
    }
    __syncthreads();

    // ---- phase 3: O += P V, wave w reduces its key slab via LDS float atomics ----
    {
        f32x8 acc[4];
#pragma unroll
        for (int t = 0; t < 4; ++t)
#pragma unroll
            for (int i = 0; i < 8; ++i) acc[t][i] = 0.f;
        int r = lane & 15;
        int koff = (lane >> 4) << 3;
#pragma unroll
        for (int ks = 0; ks < 4; ++ks) {
            int kb0 = w * 128 + ks * 32;
            bf16x16 a;
#pragma unroll
            for (int i = 0; i < 8; ++i) {
                a[i]     = (bf16)S[r][kb0 + koff + i];
                a[i + 8] = (bf16)S[r][kb0 + koff + 16 + i];
            }
#pragma unroll
            for (int t = 0; t < 4; ++t) {
                int d = t * 16 + (lane & 15);
                const bf16* vrow = vb + (size_t)d * NTOK;
                acc[t] = wmma_bf16(a, load_b_frag(vrow, kb0, lane), acc[t]);
            }
        }
        int rb = (lane >> 4) << 3;
#pragma unroll
        for (int t = 0; t < 4; ++t) {
            int d = t * 16 + (lane & 15);
#pragma unroll
            for (int j = 0; j < 8; ++j)
                atomicAdd(&Oacc[(rb + j) * DHEAD + d], acc[t][j]);
        }
    }
    __syncthreads();

    // ---- phase 4: write o[B,N,C] (c = h*64+d) as bf16 for the proj GEMM ----
#pragma unroll
    for (int i = 0; i < 4; ++i) {
        int e = tid + i * 256;                 // 0..1023
        int rr = e >> 6, d = e & 63;
        int n = qt * 16 + rr;
        o[((size_t)(b * NTOK + n)) * CDIM + h * DHEAD + d] = (bf16)Oacc[e];
    }
}

// ======================= host launch =======================
extern "C" void kernel_launch(void* const* d_in, const int* in_sizes, int n_in,
                              void* d_out, int out_size, void* d_ws, size_t ws_size,
                              hipStream_t stream) {
    const float* x     = (const float*)d_in[0];
    const float* pos   = (const float*)d_in[1];
    const float* qkv_w = (const float*)d_in[2];
    const float* qkv_b = (const float*)d_in[3];
    const float* proj_w= (const float*)d_in[4];
    const float* proj_b= (const float*)d_in[5];
    const float* ln1_g = (const float*)d_in[6];
    const float* ln1_b = (const float*)d_in[7];
    const float* ln2_g = (const float*)d_in[8];
    const float* ln2_b = (const float*)d_in[9];
    const float* fc1_w = (const float*)d_in[10];
    const float* fc1_b = (const float*)d_in[11];
    const float* fc2_w = (const float*)d_in[12];
    const float* fc2_b = (const float*)d_in[13];
    float* xout = (float*)d_out;   // residual stream lives here

    char* p = (char*)d_ws;
    auto alloc = [&](size_t bytes) {
        void* r = (void*)p;
        p += (bytes + 255) & ~(size_t)255;
        return r;
    };
    bf16* h_bf   = (bf16*)alloc((size_t)BN * CDIM * 2);
    bf16* q_bf   = (bf16*)alloc((size_t)BATCH * HEADS * NTOK * DHEAD * 2);
    bf16* k_bf   = (bf16*)alloc((size_t)BATCH * HEADS * NTOK * DHEAD * 2);
    bf16* vT_bf  = (bf16*)alloc((size_t)BATCH * HEADS * DHEAD * NTOK * 2);
    bf16* o_bf   = (bf16*)alloc((size_t)BN * CDIM * 2);
    bf16* act_bf = (bf16*)alloc((size_t)BN * FDIM * 2);
    bf16* wqT    = (bf16*)alloc((size_t)3 * CDIM * CDIM * 2);
    bf16* wpT    = (bf16*)alloc((size_t)CDIM * CDIM * 2);
    bf16* w1T    = (bf16*)alloc((size_t)FDIM * CDIM * 2);
    bf16* w2T    = (bf16*)alloc((size_t)CDIM * FDIM * 2);

    // x + pos_embed
    k_add_pos<<<dim3(BN * CDIM / 4 / 256), 256, 0, stream>>>(x, pos, xout);

    for (int l = 0; l < LAYERS; ++l) {
        const float* qw  = qkv_w  + (size_t)l * CDIM * 3 * CDIM;
        const float* qb  = qkv_b  + (size_t)l * 3 * CDIM;
        const float* pw  = proj_w + (size_t)l * CDIM * CDIM;
        const float* pb  = proj_b + (size_t)l * CDIM;
        const float* f1w = fc1_w  + (size_t)l * CDIM * FDIM;
        const float* f1b = fc1_b  + (size_t)l * FDIM;
        const float* f2w = fc2_w  + (size_t)l * FDIM * CDIM;
        const float* f2b = fc2_b  + (size_t)l * CDIM;

        // weight transpose+convert (small; all on stream, ordered)
        k_wt<<<dim3((3 * CDIM * CDIM + 255) / 256), 256, 0, stream>>>(qw,  wqT, CDIM, 3 * CDIM);
        k_wt<<<dim3((CDIM * CDIM     + 255) / 256), 256, 0, stream>>>(pw,  wpT, CDIM, CDIM);
        k_wt<<<dim3((CDIM * FDIM     + 255) / 256), 256, 0, stream>>>(f1w, w1T, CDIM, FDIM);
        k_wt<<<dim3((FDIM * CDIM     + 255) / 256), 256, 0, stream>>>(f2w, w2T, FDIM, CDIM);

        // LN1 -> h ; QKV ; attention ; proj (+residual)
        k_ln<<<dim3(BN / 4), 128, 0, stream>>>(xout, ln1_g + (size_t)l * CDIM,
                                               ln1_b + (size_t)l * CDIM, h_bf);
        k_gemm_qkv<<<dim3(3 * CDIM / 256, BN / 64), 256, 0, stream>>>(h_bf, wqT, qb,
                                                                      q_bf, k_bf, vT_bf);
        k_attn<<<dim3(NTOK / 16, HEADS, BATCH), 256, 0, stream>>>(q_bf, k_bf, vT_bf, o_bf);
        k_gemm<0><<<dim3(CDIM / 256, BN / 64), 256, 0, stream>>>(o_bf, wpT, pb,
                                                                 (void*)xout, CDIM, CDIM);

        // LN2 -> h ; FC1+GELU ; FC2 (+residual)
        k_ln<<<dim3(BN / 4), 128, 0, stream>>>(xout, ln2_g + (size_t)l * CDIM,
                                               ln2_b + (size_t)l * CDIM, h_bf);
        k_gemm<1><<<dim3(FDIM / 256, BN / 64), 256, 0, stream>>>(h_bf, w1T, f1b,
                                                                 (void*)act_bf, CDIM, FDIM);
        k_gemm<0><<<dim3(CDIM / 256, BN / 64), 256, 0, stream>>>(act_bf, w2T, f2b,
                                                                 (void*)xout, FDIM, CDIM);
    }
    (void)in_sizes; (void)n_in; (void)out_size; (void)ws_size;
}
